// RealNVP_32134945308925
// MI455X (gfx1250) — matrix operand order repaired and checked by reading, not compiled, not run on hardware
//
#include <hip/hip_runtime.h>
#include <hip/hip_bf16.h>

typedef _Float16 v16h __attribute__((ext_vector_type(16)));
typedef _Float16 v8h  __attribute__((ext_vector_type(8)));
typedef float    v8f  __attribute__((ext_vector_type(8)));
typedef float    v4f  __attribute__((ext_vector_type(4)));

union F16x16 { v16h v; v8h h[2]; };

#define LAYERS 6
#define HID    256
#define BATCH  131072
#define BM     128            // batch rows per workgroup
#define NBLK   (BATCH / BM)   // 1024 workgroups
#define MATSZ  65536          // 256*256 elements per weight matrix

// ---------------------------------------------------------------------------
// Prep: convert the 36 H x H fp32 weight matrices to f16, pre-swizzled into
// the per-lane WMMA B-fragment layout:
//   frag(kt, nt): lane L holds 16 contiguous halves j=0..15 with
//     K = kt*32 + 16*(L/16) + j ,  N = nt*16 + (L%16)
// flat ws index = ((mat*8 + kt)*16 + nt)*512 + lane*16 + j
// mat = (layer*2 + net)*3 + stage   (stage: 0=W2, 1=W3, 2=W4)
// ---------------------------------------------------------------------------
__global__ __launch_bounds__(256) void prep_weights(
    const float* __restrict__ sW2, const float* __restrict__ sW3,
    const float* __restrict__ sW4, const float* __restrict__ tW2,
    const float* __restrict__ tW3, const float* __restrict__ tW4,
    _Float16* __restrict__ wsw) {
  unsigned idx = blockIdx.x * 256u + threadIdx.x;   // 36*65536 total
  unsigned j    =  idx        & 15u;
  unsigned lane = (idx >> 4)  & 31u;
  unsigned nt   = (idx >> 9)  & 15u;
  unsigned kt   = (idx >> 13) & 7u;
  unsigned mat  =  idx >> 16;
  unsigned layer = mat / 6u;
  unsigned net   = (mat / 3u) & 1u;
  unsigned stage = mat % 3u;
  const float* W = (net == 0u)
      ? (stage == 0u ? sW2 : (stage == 1u ? sW3 : sW4))
      : (stage == 0u ? tW2 : (stage == 1u ? tW3 : tW4));
  unsigned k = kt * 32u + ((lane >> 4) << 4) + j;
  unsigned n = nt * 16u + (lane & 15u);
  wsw[idx] = (_Float16)W[layer * MATSZ + k * HID + n];
}

// ---------------------------------------------------------------------------
// Fused RealNVP inverse pass. One block = 128 batch rows through all
// 6 coupling layers x {s,t} nets. Activations live in LDS (f16, ping-pong).
// GEMM: kt-outer / mt-inner, 16 resident f32 accumulators per wave so each
// weight fragment is fetched exactly once per wave per stage; A fragments
// are software-pipelined (double-buffered) from LDS.
// ---------------------------------------------------------------------------
__global__ __launch_bounds__(256) void realnvp_fused(
    const float* __restrict__ x_in,
    const float* __restrict__ sW1, const float* __restrict__ sB1,
    const float* __restrict__ sB2, const float* __restrict__ sB3,
    const float* __restrict__ sB4, const float* __restrict__ sW5,
    const float* __restrict__ sB5,
    const float* __restrict__ tW1, const float* __restrict__ tB1,
    const float* __restrict__ tB2, const float* __restrict__ tB3,
    const float* __restrict__ tB4, const float* __restrict__ tW5,
    const float* __restrict__ tB5,
    const _Float16* __restrict__ wsw,
    float* __restrict__ out_x, float* __restrict__ out_ldi) {

  __shared__ _Float16 act[2][BM * HID];   // 2 x 64 KB ping-pong activations
  __shared__ float xRaw[BM * 2];          // async-staged raw input rows
  __shared__ float xTile[BM][2];
  __shared__ float w5s[HID];
  __shared__ float sArr[BM];
  __shared__ float tArr[BM];

  const int tid  = threadIdx.x;
  const int lane = tid & 31;
  const int wv   = tid >> 5;           // wave id 0..7
  const int bm0  = blockIdx.x * BM;

  // ---- async copy of the 256 input floats for this tile into LDS ----
  {
    unsigned ldsa = (unsigned)(uintptr_t)(&xRaw[0]) + (unsigned)tid * 4u;
    const float* g = x_in + (size_t)bm0 * 2 + tid;
    asm volatile("global_load_async_to_lds_b32 %0, %1, off"
                 :: "v"(ldsa), "v"(g) : "memory");
    asm volatile("s_wait_asynccnt 0" ::: "memory");
  }
  __syncthreads();

  float ldi = 0.0f;                    // valid for tid < BM
  if (tid < BM) {
    float x0 = xRaw[tid * 2 + 0];
    float x1 = xRaw[tid * 2 + 1];
    xTile[tid][0] = __logf(x0 / (1.0f - x0));   // logit
    xTile[tid][1] = __logf(x1 / (1.0f - x1));
  }
  __syncthreads();

  const int halfsel = lane >> 4;       // 0 or 1
  const int colsel  = lane & 15;
  const int nt0     = wv * 2;          // this wave's first n-tile
  const int rowh    = tid >> 1;        // row owned in layer-1/5 phases
  const int hsel    = tid & 1;         // column-half owned in those phases

  for (int layer = LAYERS - 1; layer >= 0; --layer) {
    const int keep = (layer & 1) ? 0 : 1;   // masked-in (kept) input dim
    const int adim = 1 - keep;              // transformed dim

    for (int net = 0; net < 2; ++net) {
      const float* W1 = net ? tW1 : sW1;
      const float* B1 = net ? tB1 : sB1;
      const float* W5 = net ? tW5 : sW5;
      const float* B5 = net ? tB5 : sB5;

      // ---- layer 1 (rank-1: only input dim `keep` is nonzero) ----
      // thread (rowh, hsel) fills 128 contiguous columns of one row:
      // W1/B1 rows streamed from global as b128, output as ds_store_b128.
      {
        const float xk = xTile[rowh][keep];
        const float* w1p = W1 + (layer * 2 + keep) * HID + hsel * 128;
        const float* b1p = B1 + layer * HID + hsel * 128;
        _Float16* orow = act[0] + rowh * HID + hsel * 128;
#pragma unroll
        for (int c = 0; c < 128; c += 8) {
          v4f wa = *(const v4f*)(w1p + c);
          v4f wb = *(const v4f*)(w1p + c + 4);
          v4f ba = *(const v4f*)(b1p + c);
          v4f bb = *(const v4f*)(b1p + c + 4);
          v8h o;
#pragma unroll
          for (int e = 0; e < 4; ++e) {
            o[e]     = (_Float16)fmaxf(xk * wa[e] + ba[e], 0.0f);
            o[e + 4] = (_Float16)fmaxf(xk * wb[e] + bb[e], 0.0f);
          }
          *(v8h*)(orow + c) = o;
        }
      }
      __syncthreads();

      // ---- three 256x256 WMMA GEMM stages with relu ----
      int bufIn = 0;
      for (int stage = 0; stage < 3; ++stage) {
        const float* Bias = net ? (stage == 0 ? tB2 : (stage == 1 ? tB3 : tB4))
                                : (stage == 0 ? sB2 : (stage == 1 ? sB3 : sB4));
        const int mat = (layer * 2 + net) * 3 + stage;
        const _Float16* wbase = wsw + (size_t)mat * MATSZ;

        // hint-prefetch next stage's weights into cache
        if (stage < 2)
          __builtin_prefetch((const void*)(wbase + MATSZ + (tid << 6)), 0, 0);

        const float b0 = Bias[layer * HID + nt0 * 16 + colsel];
        const float b1 = Bias[layer * HID + nt0 * 16 + 16 + colsel];

        const _Float16* __restrict__ in  = act[bufIn];
        _Float16* __restrict__       out = act[bufIn ^ 1];

        // 16 resident accumulators: acc[mt*2 + q] = 16x16 tile
        v8f acc[16];
#pragma unroll
        for (int i = 0; i < 16; ++i) acc[i] = (v8f){};

#pragma unroll
        for (int kt = 0; kt < 8; ++kt) {
          // this wave's two B fragments for k-step kt (loaded once)
          F16x16 b0u, b1u;
          {
            const v8h* p0 =
                (const v8h*)(wbase + (size_t)(kt * 16 + nt0) * 512 +
                             lane * 16);
            const v8h* p1 =
                (const v8h*)(wbase + (size_t)(kt * 16 + nt0 + 1) * 512 +
                             lane * 16);
            b0u.h[0] = p0[0]; b0u.h[1] = p0[1];
            b1u.h[0] = p1[0]; b1u.h[1] = p1[1];
          }
          // software-pipelined A fragments (double buffer)
          F16x16 acur, anxt;
          {
            const v8h* ap = (const v8h*)(in + colsel * HID + kt * 32 +
                                         16 * halfsel);
            acur.h[0] = ap[0];
            acur.h[1] = ap[1];
          }
#pragma unroll
          for (int mt = 0; mt < 8; ++mt) {
            if (mt < 7) {
              const v8h* ap = (const v8h*)(in + ((mt + 1) * 16 + colsel) * HID +
                                           kt * 32 + 16 * halfsel);
              anxt.h[0] = ap[0];     // K = kt*32 + 16*halfsel + [0..8)
              anxt.h[1] = ap[1];     // K = kt*32 + 16*halfsel + [8..16)
            }
            acc[mt * 2 + 0] = __builtin_amdgcn_wmma_f32_16x16x32_f16(
                false, acur.v, false, b0u.v, (short)0, acc[mt * 2 + 0], false,
                false);
            acc[mt * 2 + 1] = __builtin_amdgcn_wmma_f32_16x16x32_f16(
                false, acur.v, false, b1u.v, (short)0, acc[mt * 2 + 1], false,
                false);
            acur = anxt;
          }
        }

        // epilogue: bias + relu -> f16 activations
#pragma unroll
        for (int mt = 0; mt < 8; ++mt) {
#pragma unroll
          for (int rr = 0; rr < 8; ++rr) {
            const int row = mt * 16 + rr + 8 * halfsel;
            out[row * HID + nt0 * 16 + colsel] =
                (_Float16)fmaxf(acc[mt * 2 + 0][rr] + b0, 0.0f);
            out[row * HID + nt0 * 16 + 16 + colsel] =
                (_Float16)fmaxf(acc[mt * 2 + 1][rr] + b1, 0.0f);
          }
        }
        __syncthreads();
        bufIn ^= 1;
      }
      // final activation now in act[1]

      // ---- layer 5 (only output dim `adim` survives rm-gating) ----
      w5s[tid] = W5[(layer * HID + tid) * 2 + adim];
      __syncthreads();
      {
        float accd = 0.0f;
        const _Float16* arow = act[1] + rowh * HID + hsel * 128;
#pragma unroll 4
        for (int k = 0; k < 128; k += 8) {
          v8h v = *(const v8h*)(arow + k);
#pragma unroll
          for (int e = 0; e < 8; ++e)
            accd += (float)v[e] * w5s[hsel * 128 + k + e];
        }
        accd += __shfl_xor(accd, 1);
        if (hsel == 0) {
          float o = accd + B5[layer * 2 + adim];
          if (net == 0) sArr[rowh] = tanhf(o);
          else          tArr[rowh] = o;
        }
      }
      __syncthreads();
    }  // net

    // ---- coupling update: x_a = (x_a - t) * exp(-s);  ldi -= s ----
    if (tid < BM) {
      float s = sArr[tid];
      float t = tArr[tid];
      xTile[tid][adim] = (xTile[tid][adim] - t) * __expf(-s);
      ldi -= s;
    }
    __syncthreads();
  }  // layer

  if (tid < BM) {
    const int r = bm0 + tid;
    float x0 = xTile[tid][0];
    float x1 = xTile[tid][1];
    out_x[r * 2 + 0] = 1.0f / (1.0f + __expf(-x0));
    out_x[r * 2 + 1] = 1.0f / (1.0f + __expf(-x1));
    out_ldi[r] = ldi;
  }
}

extern "C" void kernel_launch(void* const* d_in, const int* in_sizes, int n_in,
                              void* d_out, int out_size, void* d_ws,
                              size_t ws_size, hipStream_t stream) {
  (void)in_sizes; (void)n_in; (void)out_size; (void)ws_size;
  const float* x   = (const float*)d_in[0];
  const float* sW1 = (const float*)d_in[1];
  const float* sB1 = (const float*)d_in[2];
  const float* sW2 = (const float*)d_in[3];
  const float* sB2 = (const float*)d_in[4];
  const float* sW3 = (const float*)d_in[5];
  const float* sB3 = (const float*)d_in[6];
  const float* sW4 = (const float*)d_in[7];
  const float* sB4 = (const float*)d_in[8];
  const float* sW5 = (const float*)d_in[9];
  const float* sB5 = (const float*)d_in[10];
  const float* tW1 = (const float*)d_in[11];
  const float* tB1 = (const float*)d_in[12];
  const float* tW2 = (const float*)d_in[13];
  const float* tB2 = (const float*)d_in[14];
  const float* tW3 = (const float*)d_in[15];
  const float* tB3 = (const float*)d_in[16];
  const float* tW4 = (const float*)d_in[17];
  const float* tB4 = (const float*)d_in[18];
  const float* tW5 = (const float*)d_in[19];
  const float* tB5 = (const float*)d_in[20];

  float* out_x   = (float*)d_out;
  float* out_ldi = (float*)d_out + (size_t)BATCH * 2;
  _Float16* wsw  = (_Float16*)d_ws;   // 36 * 65536 * 2 B = 4.6 MB

  // one-shot weight convert + swizzle (re-run each call: deterministic)
  prep_weights<<<(36 * MATSZ) / 256, 256, 0, stream>>>(sW2, sW3, sW4, tW2,
                                                       tW3, tW4, wsw);
  realnvp_fused<<<NBLK, 256, 0, stream>>>(
      x, sW1, sB1, sB2, sB3, sB4, sW5, sB5, tW1, tB1, tB2, tB3, tB4, tW5,
      tB5, wsw, out_x, out_ldi);
}
